// DeepSeekMoETransformerBlock_22797686407671
// MI455X (gfx1250) — compile-verified, matrix operand order checked by reading, not run on hardware
//
#include <hip/hip_runtime.h>
#include <math.h>

typedef __bf16 bf16;
typedef __attribute__((ext_vector_type(16))) __bf16 v16bf;
typedef __attribute__((ext_vector_type(8)))  float   v8f;
typedef __attribute__((ext_vector_type(4)))  unsigned int u32x4;

// Global-address-space pointer typedefs (pointee qualified with AS1).
typedef const __attribute__((address_space(1))) bf16*  gcbf;
typedef const __attribute__((address_space(1))) u32x4* gcu4;
typedef const __attribute__((address_space(1))) float* gcf;
typedef const __attribute__((address_space(1))) int*   gci;
typedef __attribute__((address_space(1))) bf16*  g_bf;
typedef __attribute__((address_space(1))) float* g_f;

// Generic->global cast via integer roundtrip (bit patterns identical on AMDGPU;
// guaranteed-to-compile form of addrspacecast).
#define TO_GLOBAL(T, p) ((T)(unsigned long long)(p))

union Frag16 { u32x4 u[2]; v16bf v; };

#define WMMA_BF16(a,b,c) __builtin_amdgcn_wmma_f32_16x16x32_bf16(false,(a),false,(b),(short)0,(c),false,false)

// LDS fragment load (ds_load_b128 x2)
static __device__ __forceinline__ v16bf ld_frag(const bf16* p0, const bf16* p1){
  Frag16 f;
  f.u[0] = *(const u32x4*)p0;
  f.u[1] = *(const u32x4*)p1;
  return f.v;
}
// Global fragment load (global_load_b128 x2)
static __device__ __forceinline__ v16bf ld_frag_g(gcbf p0, gcbf p1){
  Frag16 f;
  f.u[0] = *(gcu4)p0;
  f.u[1] = *(gcu4)p1;
  return f.v;
}

// ---------------------------------------------------------------------------
// Generic 128x128x(K) bf16 WMMA GEMM tile. A row-major [M][lda], Bt is the
// weight pre-transposed to [N][ldb] so both fragment types load contiguous.
// 8 waves: 4x2 wave grid, each wave 32x64 = 2x4 WMMA tiles.
// Register double-buffered staging: next tile's global loads are issued
// before the WMMA burst so VMEM latency overlaps matrix compute.
// ---------------------------------------------------------------------------
static __device__ void gemm_tile(gcbf A, long lda, gcbf Bt, long ldb,
                                 void* __restrict__ Cv, long ldc, gcf cin,
                                 int M, int N, int Kd, gci a_rows,
                                 int out_bf16, int nblk, int mblk,
                                 bf16* __restrict__ As, bf16* __restrict__ Bs)
{
  const int tid  = threadIdx.x;
  const int lane = tid & 31;
  const int wi   = tid >> 5;
  const int wm   = wi & 3;          // row wave 0..3
  const int wn   = wi >> 2;         // col wave 0..1
  const int r    = lane & 15;
  const int hsel = lane >> 4;

  v8f acc[2][4];
  #pragma unroll
  for (int i = 0; i < 2; i++)
    #pragma unroll
    for (int j = 0; j < 4; j++)
      #pragma unroll
      for (int t = 0; t < 8; t++) acc[i][j][t] = 0.f;

  const int krow = tid >> 2;        // 0..63
  const int kseg = tid & 3;         // 8 bf16 per seg

  // Hoisted per-row source pointers (gather index read once, not per K-tile).
  gcbf aptr[2]; gcbf bptr[2];
  bool aok[2], bok[2];
  #pragma unroll
  for (int rp = 0; rp < 2; rp++) {
    int row = krow + rp * 64;
    int m = mblk * 128 + row;
    aok[rp] = (m < M);
    long ar = 0;
    if (aok[rp]) ar = a_rows ? (long)a_rows[m] : (long)m;
    aptr[rp] = A + (size_t)ar * lda + kseg * 8;
    int n = nblk * 128 + row;
    bok[rp] = (n < N);
    long br = bok[rp] ? (long)n : 0;
    bptr[rp] = Bt + (size_t)br * ldb + kseg * 8;
  }

  const u32x4 z4 = {0u,0u,0u,0u};
  u32x4 aR[2], bR[2];
  #pragma unroll
  for (int rp = 0; rp < 2; rp++) {
    aR[rp] = aok[rp] ? *(gcu4)(aptr[rp]) : z4;
    bR[rp] = bok[rp] ? *(gcu4)(bptr[rp]) : z4;
  }

  for (int kt = 0; kt < Kd; kt += 32) {
    #pragma unroll
    for (int rp = 0; rp < 2; rp++) {
      int row = krow + rp * 64;
      *(u32x4*)(As + row * 40 + kseg * 8) = aR[rp];
      *(u32x4*)(Bs + row * 40 + kseg * 8) = bR[rp];
    }
    __syncthreads();

    // Issue next tile's global loads now; waited on only at next ds_store.
    int kt2 = kt + 32;
    if (kt2 < Kd) {
      #pragma unroll
      for (int rp = 0; rp < 2; rp++) {
        aR[rp] = aok[rp] ? *(gcu4)(aptr[rp] + kt2) : z4;
        bR[rp] = bok[rp] ? *(gcu4)(bptr[rp] + kt2) : z4;
      }
    }

    v16bf af[2], bfv[4];
    #pragma unroll
    for (int i = 0; i < 2; i++) {
      const bf16* b0 = As + (wm * 32 + i * 16 + r) * 40;
      af[i] = ld_frag(b0 + 8 * hsel, b0 + 16 + 8 * hsel);
    }
    #pragma unroll
    for (int j = 0; j < 4; j++) {
      const bf16* b0 = Bs + (wn * 64 + j * 16 + r) * 40;
      bfv[j] = ld_frag(b0 + 8 * hsel, b0 + 16 + 8 * hsel);
    }
    #pragma unroll
    for (int i = 0; i < 2; i++)
      #pragma unroll
      for (int j = 0; j < 4; j++)
        acc[i][j] = WMMA_BF16(af[i], bfv[j], acc[i][j]);
    __syncthreads();
  }

  g_bf Cb = TO_GLOBAL(g_bf, Cv);
  g_f  Cf = TO_GLOBAL(g_f,  Cv);
  #pragma unroll
  for (int i = 0; i < 2; i++) {
    #pragma unroll
    for (int j = 0; j < 4; j++) {
      #pragma unroll
      for (int t = 0; t < 8; t++) {
        int ml = wm * 32 + i * 16 + hsel * 8 + t;
        int nl = wn * 64 + j * 16 + r;
        int m = mblk * 128 + ml;
        int n = nblk * 128 + nl;
        if (m < M && n < N) {
          float v = acc[i][j][t];
          size_t o = (size_t)m * ldc + n;
          if (out_bf16) Cb[o] = (bf16)v;
          else {
            if (cin) v += cin[o];
            Cf[o] = v;
          }
        }
      }
    }
  }
}

__global__ __launch_bounds__(256) void gemm_kernel(const bf16* A, long lda, const bf16* Bt, long ldb,
    void* C, long ldc, const float* cin, int M, int N, int Kd, int out_bf16)
{
  __shared__ bf16 As[128 * 40];
  __shared__ bf16 Bs[128 * 40];
  gemm_tile(TO_GLOBAL(gcbf, A), lda, TO_GLOBAL(gcbf, Bt), ldb, C, ldc,
            TO_GLOBAL(gcf, cin), M, N, Kd, (gci) nullptr, out_bf16,
            blockIdx.x, blockIdx.y, As, Bs);
}

// MoE grouped GEMM: blockIdx.z = expert. mode 0: gate/up (A rows gathered by
// perm, bf16 out rows = slot). mode 1: down (A contiguous slot rows, f32 out).
__global__ __launch_bounds__(256) void gemm_moe_kernel(const bf16* Abase, long lda,
    const bf16* Wt, long wstride, long ldb, void* Cbase, long ldc, int N, int Kd,
    const int* offs, const int* counts, const int* perm, int mode)
{
  int e = blockIdx.z;
  int cnt = TO_GLOBAL(gci, counts)[e];
  if ((int)blockIdx.y * 128 >= cnt) return;
  int off = TO_GLOBAL(gci, offs)[e];
  __shared__ bf16 As[128 * 40];
  __shared__ bf16 Bs[128 * 40];
  gcbf A; gci a_rows; void* C; int obf;
  if (mode == 0) {
    A = TO_GLOBAL(gcbf, Abase);
    a_rows = TO_GLOBAL(gci, perm + off);
    C = (void*)((bf16*)Cbase + (size_t)off * ldc); obf = 1;
  } else {
    A = TO_GLOBAL(gcbf, Abase) + (size_t)off * lda;
    a_rows = (gci) nullptr;
    C = (void*)((float*)Cbase + (size_t)off * ldc); obf = 0;
  }
  gemm_tile(A, lda, TO_GLOBAL(gcbf, Wt) + (size_t)e * wstride, ldb, C, ldc,
            (gcf) nullptr, cnt, N, Kd, a_rows, obf, blockIdx.x, blockIdx.y, As, Bs);
}

// ---------------------------------------------------------------------------
// Elementwise / small kernels
// ---------------------------------------------------------------------------
__global__ __launch_bounds__(256) void convert_transpose_kernel(const float* __restrict__ in,
    bf16* __restrict__ out, int R, int Cc)
{
  gcf  ing  = TO_GLOBAL(gcf, in);
  g_bf outg = TO_GLOBAL(g_bf, out);
  size_t total = (size_t)R * Cc;
  size_t bbase = (size_t)blockIdx.z * total;
  size_t idx = (size_t)blockIdx.x * 256 + threadIdx.x;
  if (idx >= total) return;
  size_t r = idx / (size_t)Cc, c = idx % (size_t)Cc;
  outg[bbase + c * (size_t)R + r] = (bf16)ing[bbase + idx];
}

__global__ __launch_bounds__(256) void rmsnorm_kernel(const void* in, int in_bf16, long istride,
    const float* __restrict__ w, bf16* __restrict__ out, long ostride, int cols)
{
  int row = blockIdx.x; int tid = threadIdx.x;
  __shared__ float red[256];
  gcf  inf_ = TO_GLOBAL(gcf, in);
  gcbf inb  = TO_GLOBAL(gcbf, in);
  gcf  wg   = TO_GLOBAL(gcf, w);
  g_bf outg = TO_GLOBAL(g_bf, out);
  float ss = 0.f;
  for (int c = tid; c < cols; c += 256) {
    float v = in_bf16 ? (float)inb[(size_t)row * istride + c]
                      : inf_[(size_t)row * istride + c];
    ss += v * v;
  }
  red[tid] = ss; __syncthreads();
  for (int s = 128; s > 0; s >>= 1) { if (tid < s) red[tid] += red[tid + s]; __syncthreads(); }
  float inv = rsqrtf(red[0] / (float)cols + 1e-6f);
  for (int c = tid; c < cols; c += 256) {
    float v = in_bf16 ? (float)inb[(size_t)row * istride + c]
                      : inf_[(size_t)row * istride + c];
    outg[(size_t)row * ostride + c] = (bf16)(v * inv * wg[c]);
  }
}

// Build roped Q/K concat layouts + V from projection outputs.
__global__ __launch_bounds__(512) void prep_attn_kernel(const bf16* __restrict__ q_,
    const bf16* __restrict__ kv_, const bf16* __restrict__ ckv_,
    bf16* __restrict__ Qc_, bf16* __restrict__ Kc_, bf16* __restrict__ Vb_)
{
  const int S = 2048, HH = 16;
  const float LN1E4_32 = 0.2878231365862548f; // ln(10000)/32
  gcbf q   = TO_GLOBAL(gcbf, q_);
  gcbf kv  = TO_GLOBAL(gcbf, kv_);
  gcbf ckv = TO_GLOBAL(gcbf, ckv_);
  g_bf Qc = TO_GLOBAL(g_bf, Qc_);
  g_bf Kc = TO_GLOBAL(g_bf, Kc_);
  g_bf Vb = TO_GLOBAL(g_bf, Vb_);
  int s = blockIdx.x, h = blockIdx.y, b = blockIdx.z;
  size_t t = (size_t)b * S + s;
  size_t bhs = ((size_t)b * HH + h) * S + s;
  int j = threadIdx.x;
  float pos = (float)s;
  if (j < 192) {
    float o;
    if (j < 128) o = (float)q[t * 3072 + h * 192 + j];
    else {
      int jj = j - 128, i = jj & 31;
      float inv = __expf(-(float)i * LN1E4_32);
      float c = __cosf(pos * inv), sn = __sinf(pos * inv);
      float x0 = (float)q[t * 3072 + h * 192 + 128 + jj];
      float xr = (jj < 32) ? -(float)q[t * 3072 + h * 192 + 128 + jj + 32]
                           :  (float)q[t * 3072 + h * 192 + 128 + jj - 32];
      o = x0 * c + xr * sn;
    }
    Qc[bhs * 192 + j] = (bf16)o;
  } else if (j < 384) {
    int jk = j - 192;
    float o;
    if (jk < 128) o = (float)kv[t * 4096 + h * 256 + jk];
    else {
      int jj = jk - 128, i = jj & 31;
      float inv = __expf(-(float)i * LN1E4_32);
      float c = __cosf(pos * inv), sn = __sinf(pos * inv);
      float x0 = (float)ckv[t * 576 + 512 + jj];
      float xr = (jj < 32) ? -(float)ckv[t * 576 + 512 + jj + 32]
                           :  (float)ckv[t * 576 + 512 + jj - 32];
      o = x0 * c + xr * sn;
    }
    Kc[bhs * 192 + jk] = (bf16)o;
  } else {
    int d = j - 384;
    Vb[bhs * 128 + d] = kv[t * 4096 + h * 256 + 128 + d];
  }
}

// ---------------------------------------------------------------------------
// Flash attention: block = 8 waves, each wave owns 16 queries of a 128-query
// tile of one (b,h). K/V staged in LDS per 32-key block; online softmax done
// in WMMA C-register layout; P transposed through per-wave LDS.
// ---------------------------------------------------------------------------
__global__ __launch_bounds__(256) void flash_attn_kernel(const bf16* __restrict__ Qc_,
    const bf16* __restrict__ Kc_, const bf16* __restrict__ V_,
    bf16* __restrict__ attn_out_)
{
  const int S = 2048, HH = 16;
  const float SC = 0.07216878364870323f; // 1/sqrt(192)
  gcbf Qc = TO_GLOBAL(gcbf, Qc_);
  gcbf Kc = TO_GLOBAL(gcbf, Kc_);
  gcbf V  = TO_GLOBAL(gcbf, V_);
  g_bf attn_out = TO_GLOBAL(g_bf, attn_out_);
  __shared__ bf16 kt[32 * 200];
  __shared__ bf16 vt[128 * 40];
  __shared__ bf16 pw[8 * 16 * 32];
  int tid = threadIdx.x, wi = tid >> 5, lane = tid & 31;
  int r = lane & 15, hsel = lane >> 4;
  int qblk = blockIdx.x, h = blockIdx.y, b = blockIdx.z;
  size_t bh = (size_t)b * HH + h;
  int q0 = qblk * 128;
  int qbase_row = q0 + wi * 16;

  v16bf qf[6];
  {
    gcbf qp = Qc + (bh * S + qbase_row + r) * 192;
    #pragma unroll
    for (int kk = 0; kk < 6; kk++)
      qf[kk] = ld_frag_g(qp + kk * 32 + 8 * hsel, qp + kk * 32 + 16 + 8 * hsel);
  }
  v8f acc[8];
  #pragma unroll
  for (int j = 0; j < 8; j++)
    #pragma unroll
    for (int t = 0; t < 8; t++) acc[j][t] = 0.f;
  float mrow[8], lrow[8];
  #pragma unroll
  for (int t = 0; t < 8; t++) { mrow[t] = -INFINITY; lrow[t] = 0.f; }

  int kbmax = (q0 + 127) >> 5;
  int mymax = qbase_row + 15;

  for (int kb = 0; kb <= kbmax; ++kb) {
    { // cooperative stage of K (32x192) and V^T (128x32)
      int row = tid >> 3, c8 = tid & 7;
      gcbf kg = Kc + (bh * S + (size_t)kb * 32 + row) * 192;
      #pragma unroll
      for (int p = 0; p < 3; p++) {
        int o = (c8 + p * 8) * 8;
        *(u32x4*)(kt + row * 200 + o) = *(gcu4)(kg + o);
      }
      gcbf vg = V + (bh * S + (size_t)kb * 32 + row) * 128 + c8 * 16;
      #pragma unroll
      for (int pp = 0; pp < 2; pp++) {
        union { u32x4 u; bf16 e[8]; } cvt;
        cvt.u = *(gcu4)(vg + pp * 8);
        #pragma unroll
        for (int qq = 0; qq < 8; qq++)
          vt[(c8 * 16 + pp * 8 + qq) * 40 + row] = cvt.e[qq];
      }
    }
    __syncthreads();

    if (kb * 32 <= mymax) {            // wave-uniform: EXEC stays full
      v8f s0, s1;
      #pragma unroll
      for (int t = 0; t < 8; t++) { s0[t] = 0.f; s1[t] = 0.f; }
      #pragma unroll
      for (int kk = 0; kk < 6; kk++) {
        const bf16* k0p = kt + r * 200 + kk * 32;
        const bf16* k1p = kt + (16 + r) * 200 + kk * 32;
        v16bf b0 = ld_frag(k0p + 8 * hsel, k0p + 16 + 8 * hsel);
        v16bf b1 = ld_frag(k1p + 8 * hsel, k1p + 16 + 8 * hsel);
        s0 = WMMA_BF16(qf[kk], b0, s0);
        s1 = WMMA_BF16(qf[kk], b1, s1);
      }
      float fr[8];
      bf16* pwv = pw + wi * 512;
      #pragma unroll
      for (int t = 0; t < 8; t++) {
        int qr = qbase_row + hsel * 8 + t;
        int k0i = kb * 32 + r;
        int k1i = k0i + 16;
        float a0 = (k0i <= qr) ? s0[t] * SC : -INFINITY;
        float a1 = (k1i <= qr) ? s1[t] * SC : -INFINITY;
        float mx = fmaxf(a0, a1);
        #pragma unroll
        for (int d = 1; d < 16; d <<= 1) mx = fmaxf(mx, __shfl_xor(mx, d, 32));
        float mnew = fmaxf(mrow[t], mx);
        float p0 = __expf(a0 - mnew);
        float p1 = __expf(a1 - mnew);
        float rs = p0 + p1;
        #pragma unroll
        for (int d = 1; d < 16; d <<= 1) rs += __shfl_xor(rs, d, 32);
        fr[t] = __expf(mrow[t] - mnew);
        mrow[t] = mnew;
        lrow[t] = lrow[t] * fr[t] + rs;
        pwv[(hsel * 8 + t) * 32 + r] = (bf16)p0;
        pwv[(hsel * 8 + t) * 32 + 16 + r] = (bf16)p1;
      }
      #pragma unroll
      for (int j = 0; j < 8; j++)
        #pragma unroll
        for (int t = 0; t < 8; t++) acc[j][t] *= fr[t];
      asm volatile("s_wait_dscnt 0" ::: "memory");   // intra-wave P transpose hazard
      const bf16* pp0 = pw + wi * 512 + r * 32;
      v16bf pf = ld_frag(pp0 + 8 * hsel, pp0 + 16 + 8 * hsel);
      #pragma unroll
      for (int j = 0; j < 8; j++) {
        const bf16* vb0 = vt + (j * 16 + r) * 40;
        v16bf vf = ld_frag(vb0 + 8 * hsel, vb0 + 16 + 8 * hsel);
        acc[j] = WMMA_BF16(pf, vf, acc[j]);
      }
    }
    __syncthreads();
  }
  #pragma unroll
  for (int j = 0; j < 8; j++) {
    #pragma unroll
    for (int t = 0; t < 8; t++) {
      int qr = qbase_row + hsel * 8 + t;
      size_t token = (size_t)b * S + qr;
      attn_out[token * 2048 + h * 128 + j * 16 + r] = (bf16)(acc[j][t] / lrow[t]);
    }
  }
}

// ---------------------------------------------------------------------------
// Router / MoE plumbing
// ---------------------------------------------------------------------------
__global__ void zero_counts_kernel(int* counts)
{
  if (threadIdx.x < 8) counts[threadIdx.x] = 0;
}

__global__ __launch_bounds__(256) void router_kernel(const bf16* __restrict__ h2_,
    const float* __restrict__ wr_, float* __restrict__ probs,
    float* __restrict__ topw, int* __restrict__ topi, int* __restrict__ counts)
{
  gcbf h2 = TO_GLOBAL(gcbf, h2_);
  gcf  wr = TO_GLOBAL(gcf, wr_);
  int t = blockIdx.x; int tid = threadIdx.x;
  float lg[8] = {0,0,0,0,0,0,0,0};
  for (int d = tid; d < 1024; d += 256) {
    float hv = (float)h2[(size_t)t * 1024 + d];
    #pragma unroll
    for (int e = 0; e < 8; e++) lg[e] += hv * wr[d * 8 + e];
  }
  __shared__ float red[8 * 256];
  #pragma unroll
  for (int e = 0; e < 8; e++) red[e * 256 + tid] = lg[e];
  __syncthreads();
  for (int s = 128; s > 0; s >>= 1) {
    if (tid < s)
      #pragma unroll
      for (int e = 0; e < 8; e++) red[e * 256 + tid] += red[e * 256 + tid + s];
    __syncthreads();
  }
  if (tid == 0) {
    float l[8], p[8];
    float m = red[0];
    for (int e = 0; e < 8; e++) { l[e] = red[e * 256]; m = fmaxf(m, l[e]); }
    float sum = 0.f;
    for (int e = 0; e < 8; e++) { p[e] = __expf(l[e] - m); sum += p[e]; }
    for (int e = 0; e < 8; e++) { p[e] /= sum; probs[(size_t)t * 8 + e] = p[e]; }
    int i0 = 0;
    for (int e = 1; e < 8; e++) if (p[e] > p[i0]) i0 = e;
    int i1 = -1;
    for (int e = 0; e < 8; e++) if (e != i0 && (i1 < 0 || p[e] > p[i1])) i1 = e;
    float w0 = p[i0], w1 = p[i1], nrm = w0 + w1;
    topw[t * 2] = w0 / nrm; topw[t * 2 + 1] = w1 / nrm;
    topi[t * 2] = i0; topi[t * 2 + 1] = i1;
    atomicAdd(&counts[i0], 1); atomicAdd(&counts[i1], 1);
  }
}

// Deterministic expert permutation: thread e scans tokens in order.
__global__ void scatter_kernel(const int* __restrict__ topi, const int* __restrict__ counts,
                               int* __restrict__ offs, int* __restrict__ perm,
                               int* __restrict__ slot_of)
{
  __shared__ int so[9];
  if (threadIdx.x == 0) {
    so[0] = 0;
    for (int e = 0; e < 8; e++) so[e + 1] = so[e] + counts[e];
    for (int e = 0; e < 9; e++) offs[e] = so[e];
  }
  __syncthreads();
  int e = threadIdx.x;
  if (e < 8) {
    int pos = so[e];
    for (int t = 0; t < 4096; t++) {
      int i0 = topi[t * 2], i1 = topi[t * 2 + 1];
      if (i0 == e)      { perm[pos] = t; slot_of[t * 2] = pos; pos++; }
      else if (i1 == e) { perm[pos] = t; slot_of[t * 2 + 1] = pos; pos++; }
    }
  }
}

__global__ __launch_bounds__(256) void silu_mul_kernel(bf16* __restrict__ g_, const bf16* __restrict__ u_)
{
  g_bf g = TO_GLOBAL(g_bf, g_);
  gcbf u = TO_GLOBAL(gcbf, u_);
  size_t idx = (size_t)blockIdx.x * 256 + threadIdx.x;
  if (idx >= (size_t)8192 * 4096) return;
  float gv = (float)g[idx], uv = (float)u[idx];
  g[idx] = (bf16)((gv / (1.f + __expf(-gv))) * uv);
}

__global__ __launch_bounds__(256) void combine_kernel(const float* __restrict__ xmid_,
    const float* __restrict__ ffs_, const int* __restrict__ slot_of_,
    const float* __restrict__ topw_, float* __restrict__ out_)
{
  gcf xmid = TO_GLOBAL(gcf, xmid_);
  gcf ffs  = TO_GLOBAL(gcf, ffs_);
  gci slot_of = TO_GLOBAL(gci, slot_of_);
  gcf topw = TO_GLOBAL(gcf, topw_);
  g_f out  = TO_GLOBAL(g_f, out_);
  size_t idx = (size_t)blockIdx.x * 256 + threadIdx.x;
  if (idx >= (size_t)4096 * 1024) return;
  size_t t = idx >> 10;
  int d = (int)(idx & 1023);
  int s0 = slot_of[t * 2], s1 = slot_of[t * 2 + 1];
  out[idx] = xmid[idx] + topw[t * 2] * ffs[(size_t)s0 * 1024 + d]
                       + topw[t * 2 + 1] * ffs[(size_t)s1 * 1024 + d];
}

__global__ __launch_bounds__(256) void aux_kernel(const float* __restrict__ probs,
                                                  const int* __restrict__ counts,
                                                  float* __restrict__ out)
{
  __shared__ float red[256];
  __shared__ float Pe[8];
  int tid = threadIdx.x;
  float part[8] = {0,0,0,0,0,0,0,0};
  for (int t = tid; t < 4096; t += 256)
    #pragma unroll
    for (int e = 0; e < 8; e++) part[e] += probs[(size_t)t * 8 + e];
  for (int e = 0; e < 8; e++) {
    red[tid] = part[e]; __syncthreads();
    for (int s = 128; s > 0; s >>= 1) { if (tid < s) red[tid] += red[tid + s]; __syncthreads(); }
    if (tid == 0) Pe[e] = red[0] / 4096.f;
    __syncthreads();
  }
  if (tid == 0) {
    float a = 0.f;
    for (int e = 0; e < 8; e++) a += ((float)counts[e] / 8192.f) * Pe[e];
    out[0] = 8.f * a;
  }
}

// ---------------------------------------------------------------------------
extern "C" void kernel_launch(void* const* d_in, const int* in_sizes, int n_in,
                              void* d_out, int out_size, void* d_ws, size_t ws_size,
                              hipStream_t stream)
{
  const float* x    = (const float*)d_in[0];
  const float* ln1w = (const float*)d_in[1];
  const float* ln2w = (const float*)d_in[2];
  const float* wq   = (const float*)d_in[3];
  const float* wkva = (const float*)d_in[4];
  const float* kvnw = (const float*)d_in[5];
  const float* wkvb = (const float*)d_in[6];
  const float* wo   = (const float*)d_in[7];
  const float* wr   = (const float*)d_in[8];
  const float* wg   = (const float*)d_in[9];
  const float* wu   = (const float*)d_in[10];
  const float* wd   = (const float*)d_in[11];
  float* out = (float*)d_out;

  const int T = 4096, D = 1024, S = 2048, Bb = 2, Hh = 16, F = 4096, Ee = 8;

  char* base = (char*)d_ws;
  size_t off = 0;
  auto bump = [&](size_t bytes) -> char* {
    char* p = base + off;
    off = (off + bytes + 255) & ~(size_t)255;
    return p;
  };

  // ---- Region X: attention-phase buffers (later aliased by gate/up) ----
  bf16* q_b    = (bf16*)bump((size_t)T * 3072 * 2);
  bf16* ckv_b  = (bf16*)bump((size_t)T * 576 * 2);
  bf16* kv_b   = (bf16*)bump((size_t)T * 4096 * 2);
  bf16* Qcat   = (bf16*)bump((size_t)Bb * Hh * S * 192 * 2);
  bf16* Kcat   = (bf16*)bump((size_t)Bb * Hh * S * 192 * 2);
  bf16* Vb     = (bf16*)bump((size_t)Bb * Hh * S * 128 * 2);
  bf16* attn_b = (bf16*)bump((size_t)T * 2048 * 2);
  // MoE-phase aliases into Region X (147 MB >= 2 x 64 MB needed)
  bf16* g_b = (bf16*)d_ws;
  bf16* u_b = (bf16*)((char*)d_ws + (size_t)2 * T * F * 2);

  // ---- Persistent buffers ----
  bf16* h_b     = (bf16*)bump((size_t)T * D * 2);
  bf16* ckvn_b  = (bf16*)bump((size_t)T * 512 * 2);
  bf16* wq_t    = (bf16*)bump((size_t)3072 * 1024 * 2);
  bf16* wkva_t  = (bf16*)bump((size_t)576 * 1024 * 2);
  bf16* wkvb_t  = (bf16*)bump((size_t)4096 * 512 * 2);
  bf16* wo_t    = (bf16*)bump((size_t)1024 * 2048 * 2);
  bf16* wg_t    = (bf16*)bump((size_t)Ee * F * D * 2);
  bf16* wu_t    = (bf16*)bump((size_t)Ee * F * D * 2);
  bf16* wd_t    = (bf16*)bump((size_t)Ee * D * F * 2);
  float* xmid   = (float*)bump((size_t)T * D * 4);
  bf16* h2_b    = (bf16*)bump((size_t)T * D * 2);
  float* probs  = (float*)bump((size_t)T * 8 * 4);
  float* topw   = (float*)bump((size_t)T * 2 * 4);
  int*   topi   = (int*)bump((size_t)T * 2 * 4);
  int*   counts = (int*)bump(64);
  int*   offs   = (int*)bump(64);
  int*   perm   = (int*)bump((size_t)2 * T * 4);
  int*   slot_of= (int*)bump((size_t)2 * T * 4);
  float* ffs    = (float*)bump((size_t)2 * T * D * 4);

  dim3 blk(256);

  // Weight convert + transpose to bf16 [N][K]
  {
    size_t tot;
    tot = (size_t)1024 * 3072;
    convert_transpose_kernel<<<dim3((unsigned)((tot + 255) / 256), 1, 1), blk, 0, stream>>>(wq, wq_t, 1024, 3072);
    tot = (size_t)1024 * 576;
    convert_transpose_kernel<<<dim3((unsigned)((tot + 255) / 256), 1, 1), blk, 0, stream>>>(wkva, wkva_t, 1024, 576);
    tot = (size_t)512 * 4096;
    convert_transpose_kernel<<<dim3((unsigned)((tot + 255) / 256), 1, 1), blk, 0, stream>>>(wkvb, wkvb_t, 512, 4096);
    tot = (size_t)2048 * 1024;
    convert_transpose_kernel<<<dim3((unsigned)((tot + 255) / 256), 1, 1), blk, 0, stream>>>(wo, wo_t, 2048, 1024);
    tot = (size_t)1024 * 4096;
    convert_transpose_kernel<<<dim3((unsigned)((tot + 255) / 256), 1, 8), blk, 0, stream>>>(wg, wg_t, 1024, 4096);
    convert_transpose_kernel<<<dim3((unsigned)((tot + 255) / 256), 1, 8), blk, 0, stream>>>(wu, wu_t, 1024, 4096);
    tot = (size_t)4096 * 1024;
    convert_transpose_kernel<<<dim3((unsigned)((tot + 255) / 256), 1, 8), blk, 0, stream>>>(wd, wd_t, 4096, 1024);
  }

  // Attention path
  rmsnorm_kernel<<<T, blk, 0, stream>>>((const void*)x, 0, 1024, ln1w, h_b, 1024, 1024);
  gemm_kernel<<<dim3(24, 32), blk, 0, stream>>>(h_b, 1024, wq_t, 1024, q_b, 3072, nullptr, T, 3072, 1024, 1);
  gemm_kernel<<<dim3(5, 32), blk, 0, stream>>>(h_b, 1024, wkva_t, 1024, ckv_b, 576, nullptr, T, 576, 1024, 1);
  rmsnorm_kernel<<<T, blk, 0, stream>>>((const void*)ckv_b, 1, 576, kvnw, ckvn_b, 512, 512);
  gemm_kernel<<<dim3(32, 32), blk, 0, stream>>>(ckvn_b, 512, wkvb_t, 512, kv_b, 4096, nullptr, T, 4096, 512, 1);
  prep_attn_kernel<<<dim3(S, Hh, Bb), dim3(512), 0, stream>>>(q_b, kv_b, ckv_b, Qcat, Kcat, Vb);
  flash_attn_kernel<<<dim3(16, Hh, Bb), blk, 0, stream>>>(Qcat, Kcat, Vb, attn_b);
  gemm_kernel<<<dim3(8, 32), blk, 0, stream>>>(attn_b, 2048, wo_t, 2048, xmid, 1024, x, T, 1024, 2048, 0);

  // MoE path
  rmsnorm_kernel<<<T, blk, 0, stream>>>((const void*)xmid, 0, 1024, ln2w, h2_b, 1024, 1024);
  zero_counts_kernel<<<1, 32, 0, stream>>>(counts);
  router_kernel<<<T, blk, 0, stream>>>(h2_b, wr, probs, topw, topi, counts);
  scatter_kernel<<<1, 32, 0, stream>>>(topi, counts, offs, perm, slot_of);
  gemm_moe_kernel<<<dim3(32, 32, 8), blk, 0, stream>>>(h2_b, 1024, wg_t, (long)F * D, 1024,
      (void*)g_b, 4096, 4096, 1024, offs, counts, perm, 0);
  gemm_moe_kernel<<<dim3(32, 32, 8), blk, 0, stream>>>(h2_b, 1024, wu_t, (long)F * D, 1024,
      (void*)u_b, 4096, 4096, 1024, offs, counts, perm, 0);
  silu_mul_kernel<<<dim3(131072), blk, 0, stream>>>(g_b, u_b);
  gemm_moe_kernel<<<dim3(8, 32, 8), blk, 0, stream>>>(g_b, 4096, wd_t, (long)D * F, 4096,
      (void*)ffs, 1024, 1024, 4096, offs, counts, perm, 1);
  combine_kernel<<<dim3(16384), blk, 0, stream>>>(xmid, ffs, slot_of, topw, out);
  aux_kernel<<<1, blk, 0, stream>>>(probs, counts, out + (size_t)T * D);
}
